// MHC_GNN_82721070121703
// MI455X (gfx1250) — matrix operand order repaired and checked by reading, not compiled, not run on hardware
//
#include <hip/hip_runtime.h>
#include <hip/hip_bf16.h>

// ---------------------------------------------------------------------------
// MHC-GNN for MI455X (gfx1250, wave32, WMMA bf16 w/ fp32 accumulate)
// ---------------------------------------------------------------------------

#define NN   131072   // nodes
#define DD   128      // hidden dim
#define DX   64       // raw feature dim
#define NG   1024     // graphs
#define EE   4194304  // edges per hop
#define HH   4        // hops per layer
#define TILES (NN / 16)
#define ZSTRIDE 136   // 128 + 8 bf16 pad to break LDS bank conflicts

typedef __attribute__((ext_vector_type(16))) __bf16 v16bf;
typedef __attribute__((ext_vector_type(8)))  __bf16 v8bf;
typedef __attribute__((ext_vector_type(8)))  float  v8f;

// Pack 16 f32 values (two 8-wide K-groups) into a bf16 A-fragment register set.
static __device__ inline v16bf pack_a(float4 f0, float4 f1, float4 g0, float4 g1) {
  v16bf a;
  a[0]  = (__bf16)f0.x; a[1]  = (__bf16)f0.y; a[2]  = (__bf16)f0.z; a[3]  = (__bf16)f0.w;
  a[4]  = (__bf16)f1.x; a[5]  = (__bf16)f1.y; a[6]  = (__bf16)f1.z; a[7]  = (__bf16)f1.w;
  a[8]  = (__bf16)g0.x; a[9]  = (__bf16)g0.y; a[10] = (__bf16)g0.z; a[11] = (__bf16)g0.w;
  a[12] = (__bf16)g1.x; a[13] = (__bf16)g1.y; a[14] = (__bf16)g1.z; a[15] = (__bf16)g1.w;
  return a;
}

// ---------------------------------------------------------------------------
// Weight transpose + f32 -> bf16: dst[h][n][k] = src[h][k][n]
// (so each lane's B-fragment is two 16-byte runs of W^T row n, 16 k apart)
// ---------------------------------------------------------------------------
__global__ void transpose_bf16_kernel(const float* __restrict__ src,
                                      __bf16* __restrict__ dst,
                                      int Hn, int K, int Nn) {
  int idx = blockIdx.x * blockDim.x + threadIdx.x;
  int total = Hn * K * Nn;
  if (idx >= total) return;
  int n = idx % Nn;
  int t = idx / Nn;
  int k = t % K;
  int h = t / K;
  dst[((size_t)h * Nn + n) * K + k] = (__bf16)src[idx];
}

__global__ void zero_kernel(float* __restrict__ p, size_t n) {
  size_t i = (size_t)blockIdx.x * blockDim.x + threadIdx.x;
  size_t stride = (size_t)gridDim.x * blockDim.x;
  for (; i < n; i += stride) p[i] = 0.0f;
}

// ---------------------------------------------------------------------------
// One MHC layer: out[n,:] = sum_h ( relu(X_h W1_h + b1_h) W2_h + b2_h )
// Each wave computes a 16-row tile; 256 v_wmma_f32_16x16x32_bf16 per tile.
// B fragments are preloaded per n-tile into a register array so the loads
// clause together and WMMAs issue back-to-back (one wait per n-tile, not 4).
// SPLIT=true: hop0 input is concat(x[:, :64], rw[:, :64]).
// ---------------------------------------------------------------------------
template <bool SPLIT>
__global__ __launch_bounds__(128) void mhc_layer_kernel(
    const float* __restrict__ h0a, const float* __restrict__ h0b,
    const float* __restrict__ h1,  const float* __restrict__ h2,
    const float* __restrict__ h3,
    const __bf16* __restrict__ Wt1, const float* __restrict__ B1,
    const __bf16* __restrict__ Wt2, const float* __restrict__ B2,
    float* __restrict__ out) {
  __shared__ __bf16 lds_z[4][16 * ZSTRIDE];
  const int wave = threadIdx.x >> 5;
  const int lane = threadIdx.x & 31;
  const int tile = blockIdx.x * 4 + wave;
  const int m    = lane & 15;   // A row within tile / C column-within-ntile
  const int half = lane >> 4;
  const int row  = tile * 16 + m;
  __bf16* zt = &lds_z[wave][0];

  // y accumulators; fold sum_h b2[h][n] into the init.
  v8f y[8];
#pragma unroll
  for (int nt = 0; nt < 8; ++nt) {
    int n = nt * 16 + m;
    float bs = B2[0 * DD + n] + B2[1 * DD + n] + B2[2 * DD + n] + B2[3 * DD + n];
#pragma unroll
    for (int i = 0; i < 8; ++i) y[nt][i] = bs;
  }

  for (int h = 0; h < HH; ++h) {
    // ---- A fragments for X_h: 16x128 tile, f32 global -> bf16 regs ----
    v16bf af[4];
#pragma unroll
    for (int kc = 0; kc < 4; ++kc) {
      int s0 = kc * 32 + half * 8;
      int s1 = s0 + 16;
      const float *p0, *p1;
      if (SPLIT && h == 0) {
        p0 = (s0 < 64) ? (h0a + (size_t)row * DX + s0)
                       : (h0b + (size_t)row * DX + (s0 - 64));
        p1 = (s1 < 64) ? (h0a + (size_t)row * DX + s1)
                       : (h0b + (size_t)row * DX + (s1 - 64));
      } else {
        const float* base = (h == 0) ? h0a : (h == 1) ? h1 : (h == 2) ? h2 : h3;
        p0 = base + (size_t)row * DD + s0;
        p1 = p0 + 16;
      }
      af[kc] = pack_a(*(const float4*)p0, *(const float4*)(p0 + 4),
                      *(const float4*)p1, *(const float4*)(p1 + 4));
    }

    const __bf16* W1h = Wt1 + (size_t)h * DD * DD;
    const __bf16* W2h = Wt2 + (size_t)h * DD * DD;

    // ---- z = relu(X W1 + b1), stored to LDS as bf16 (C-layout -> row major)
#pragma unroll
    for (int nt = 0; nt < 8; ++nt) {
      int n = nt * 16 + m;
      v16bf barr[4];
#pragma unroll
      for (int kc = 0; kc < 4; ++kc)
        barr[kc] = __builtin_shufflevector(*(const v8bf*)(W1h + (size_t)n * DD + kc * 32 + half * 8), *(const v8bf*)(W1h + (size_t)n * DD + kc * 32 + half * 8 + 16), 0, 1, 2, 3, 4, 5, 6, 7, 8, 9, 10, 11, 12, 13, 14, 15);
      v8f c = {};
#pragma unroll
      for (int kc = 0; kc < 4; ++kc)
        c = __builtin_amdgcn_wmma_f32_16x16x32_bf16(false, af[kc], false, barr[kc],
                                                    (short)0, c, false, false);
      float bb = B1[h * DD + n];
#pragma unroll
      for (int r = 0; r < 8; ++r) {
        float v = c[r] + bb;
        v = v > 0.0f ? v : 0.0f;
        zt[(half * 8 + r) * ZSTRIDE + n] = (__bf16)v;
      }
    }
    __syncthreads();

    // ---- rebuild z as A fragments from LDS (transpose round trip) ----
    v16bf zf[4];
#pragma unroll
    for (int kc = 0; kc < 4; ++kc) {
      int s0 = kc * 32 + half * 8;
      const v8bf lo = *(const v8bf*)(zt + m * ZSTRIDE + s0);
      const v8bf hi = *(const v8bf*)(zt + m * ZSTRIDE + s0 + 16);
      v16bf a;
#pragma unroll
      for (int i = 0; i < 8; ++i) { a[i] = lo[i]; a[8 + i] = hi[i]; }
      zf[kc] = a;
    }
    __syncthreads();

    // ---- y += z W2 ----
#pragma unroll
    for (int nt = 0; nt < 8; ++nt) {
      int n = nt * 16 + m;
      v16bf barr[4];
#pragma unroll
      for (int kc = 0; kc < 4; ++kc)
        barr[kc] = __builtin_shufflevector(*(const v8bf*)(W2h + (size_t)n * DD + kc * 32 + half * 8), *(const v8bf*)(W2h + (size_t)n * DD + kc * 32 + half * 8 + 16), 0, 1, 2, 3, 4, 5, 6, 7, 8, 9, 10, 11, 12, 13, 14, 15);
#pragma unroll
      for (int kc = 0; kc < 4; ++kc)
        y[nt] = __builtin_amdgcn_wmma_f32_16x16x32_bf16(false, zf[kc], false, barr[kc],
                                                        (short)0, y[nt], false, false);
    }
  }

  // ---- store: C layout -> out[row][col] (f32) ----
#pragma unroll
  for (int nt = 0; nt < 8; ++nt) {
    int n = nt * 16 + m;
#pragma unroll
    for (int r = 0; r < 8; ++r) {
      out[(size_t)(tile * 16 + half * 8 + r) * DD + n] = y[nt][r];
    }
  }
}

// ---------------------------------------------------------------------------
// SPMM: out[r,:] += x[c,:] per edge. One wave per edge, 4 floats per lane.
// Edge id is made explicitly wave-uniform (readfirstlane) so index loads
// become scalar and the row gather uses the SADDR form. A one-iteration
// look-ahead prefetch (global_prefetch_b8) hides gather latency.
// ---------------------------------------------------------------------------
__global__ __launch_bounds__(256) void spmm_kernel(
    const int* __restrict__ rows, const int* __restrict__ cols,
    const float* __restrict__ x, float* __restrict__ out, int nedges) {
  int wid  = (blockIdx.x * blockDim.x + threadIdx.x) >> 5;
  int lane = threadIdx.x & 31;
  int nw   = (gridDim.x * blockDim.x) >> 5;
  for (int e0 = wid; e0 < nedges; e0 += nw) {
    int e = __builtin_amdgcn_readfirstlane(e0);   // wave-uniform -> scalar loads
    int r = rows[e];
    int c = cols[e];
    // Prefetch next iteration's source row while this one is in flight.
    int en = e + nw;
    if (en < nedges) {
      int cn = cols[en];
      __builtin_prefetch(&x[(size_t)cn * DD + lane * 4], 0, 0);
    }
    float4 v = *(const float4*)&x[(size_t)c * DD + lane * 4];
    float* dst = &out[(size_t)r * DD + lane * 4];
    __hip_atomic_fetch_add(dst + 0, v.x, __ATOMIC_RELAXED, __HIP_MEMORY_SCOPE_AGENT);
    __hip_atomic_fetch_add(dst + 1, v.y, __ATOMIC_RELAXED, __HIP_MEMORY_SCOPE_AGENT);
    __hip_atomic_fetch_add(dst + 2, v.z, __ATOMIC_RELAXED, __HIP_MEMORY_SCOPE_AGENT);
    __hip_atomic_fetch_add(dst + 3, v.w, __ATOMIC_RELAXED, __HIP_MEMORY_SCOPE_AGENT);
  }
}

// ---------------------------------------------------------------------------
// Final: out_node = concat(x1,x2) @ lin_w + lin_b, atomically pooled by batch.
// K = 256 -> 8 WMMA k-chunks per n-tile; B fragments preloaded per n-tile.
// ---------------------------------------------------------------------------
__global__ __launch_bounds__(128) void final_pool_kernel(
    const float* __restrict__ x1, const float* __restrict__ x2,
    const __bf16* __restrict__ Wt, const float* __restrict__ bias,
    const int* __restrict__ batch, float* __restrict__ out) {
  const int wave = threadIdx.x >> 5;
  const int lane = threadIdx.x & 31;
  const int tile = blockIdx.x * 4 + wave;
  const int m    = lane & 15;
  const int half = lane >> 4;
  const int row  = tile * 16 + m;

  v16bf af[8];
#pragma unroll
  for (int kc = 0; kc < 8; ++kc) {
    int s0 = kc * 32 + half * 8;
    int s1 = s0 + 16;
    const float* p0 = (s0 < DD) ? (x1 + (size_t)row * DD + s0)
                                : (x2 + (size_t)row * DD + (s0 - DD));
    const float* p1 = (s1 < DD) ? (x1 + (size_t)row * DD + s1)
                                : (x2 + (size_t)row * DD + (s1 - DD));
    af[kc] = pack_a(*(const float4*)p0, *(const float4*)(p0 + 4),
                    *(const float4*)p1, *(const float4*)(p1 + 4));
  }

  int g[8];
#pragma unroll
  for (int r = 0; r < 8; ++r) g[r] = batch[tile * 16 + half * 8 + r];

#pragma unroll
  for (int nt = 0; nt < 8; ++nt) {
    int n = nt * 16 + m;
    v16bf barr[8];
#pragma unroll
    for (int kc = 0; kc < 8; ++kc)
      barr[kc] = __builtin_shufflevector(*(const v8bf*)(Wt + (size_t)n * 256 + kc * 32 + half * 8), *(const v8bf*)(Wt + (size_t)n * 256 + kc * 32 + half * 8 + 16), 0, 1, 2, 3, 4, 5, 6, 7, 8, 9, 10, 11, 12, 13, 14, 15);
    float bb = bias[n];
    v8f c;
#pragma unroll
    for (int i = 0; i < 8; ++i) c[i] = bb;
#pragma unroll
    for (int kc = 0; kc < 8; ++kc)
      c = __builtin_amdgcn_wmma_f32_16x16x32_bf16(false, af[kc], false, barr[kc],
                                                  (short)0, c, false, false);
#pragma unroll
    for (int r = 0; r < 8; ++r) {
      __hip_atomic_fetch_add(&out[(size_t)g[r] * DD + n], c[r],
                             __ATOMIC_RELAXED, __HIP_MEMORY_SCOPE_AGENT);
    }
  }
}

// ---------------------------------------------------------------------------
// Launch
// ---------------------------------------------------------------------------
extern "C" void kernel_launch(void* const* d_in, const int* in_sizes, int n_in,
                              void* d_out, int out_size, void* d_ws, size_t ws_size,
                              hipStream_t stream) {
  (void)in_sizes; (void)n_in; (void)out_size; (void)ws_size;

  const float* x    = (const float*)d_in[0];
  const float* rw   = (const float*)d_in[1];
  const float* h1f  = (const float*)d_in[2];
  const float* h2f  = (const float*)d_in[3];
  const float* h3f  = (const float*)d_in[4];
  const int*   e1   = (const int*)d_in[5];
  const int*   e2   = (const int*)d_in[6];
  const int*   e3   = (const int*)d_in[7];
  const int*   batch= (const int*)d_in[8];
  const float* l1w1 = (const float*)d_in[9];
  const float* l1b1 = (const float*)d_in[10];
  const float* l1w2 = (const float*)d_in[11];
  const float* l1b2 = (const float*)d_in[12];
  const float* l2w1 = (const float*)d_in[13];
  const float* l2b1 = (const float*)d_in[14];
  const float* l2w2 = (const float*)d_in[15];
  const float* l2b2 = (const float*)d_in[16];
  const float* linw = (const float*)d_in[17];
  const float* linb = (const float*)d_in[18];
  float* out = (float*)d_out;

  // Workspace layout (bf16 weights in first MB, then 5 x 64MB f32 buffers).
  char* ws = (char*)d_ws;
  __bf16* wt_l1_1 = (__bf16*)(ws + 0);
  __bf16* wt_l1_2 = (__bf16*)(ws + 131072);
  __bf16* wt_l2_1 = (__bf16*)(ws + 262144);
  __bf16* wt_l2_2 = (__bf16*)(ws + 393216);
  __bf16* lin_wt  = (__bf16*)(ws + 524288);
  float*  x1 = (float*)(ws + (1u << 20));
  float*  x2 = x1 + (size_t)NN * DD;
  float*  s1 = x2 + (size_t)NN * DD;
  float*  s2 = s1 + (size_t)NN * DD;
  float*  s3 = s2 + (size_t)NN * DD;

  // Weight transpose + bf16 conversion (tiny).
  {
    int total = HH * DD * DD;
    int blk = (total + 255) / 256;
    transpose_bf16_kernel<<<blk, 256, 0, stream>>>(l1w1, wt_l1_1, HH, DD, DD);
    transpose_bf16_kernel<<<blk, 256, 0, stream>>>(l1w2, wt_l1_2, HH, DD, DD);
    transpose_bf16_kernel<<<blk, 256, 0, stream>>>(l2w1, wt_l2_1, HH, DD, DD);
    transpose_bf16_kernel<<<blk, 256, 0, stream>>>(l2w2, wt_l2_2, HH, DD, DD);
    int tl = 2 * DD * DD;
    transpose_bf16_kernel<<<(tl + 255) / 256, 256, 0, stream>>>(linw, lin_wt, 1, 2 * DD, DD);
  }

  // Zero atomic targets (s1..s3 contiguous, plus pooled output).
  zero_kernel<<<4096, 256, 0, stream>>>(s1, (size_t)3 * NN * DD);
  zero_kernel<<<256, 256, 0, stream>>>(out, (size_t)NG * DD);

  // Layer 1 (hop0 = concat(x, rw)).
  mhc_layer_kernel<true><<<TILES / 4, 128, 0, stream>>>(
      x, rw, h1f, h2f, h3f, wt_l1_1, l1b1, wt_l1_2, l1b2, x1);

  // SPMM per hop (edges stored [2, E]: rows then cols).
  spmm_kernel<<<8192, 256, 0, stream>>>(e1, e1 + EE, x1, s1, EE);
  spmm_kernel<<<8192, 256, 0, stream>>>(e2, e2 + EE, x1, s2, EE);
  spmm_kernel<<<8192, 256, 0, stream>>>(e3, e3 + EE, x1, s3, EE);

  // Layer 2.
  mhc_layer_kernel<false><<<TILES / 4, 128, 0, stream>>>(
      x1, nullptr, s1, s2, s3, wt_l2_1, l2b1, wt_l2_2, l2b2, x2);

  // Final linear + pooled segment sum.
  final_pool_kernel<<<TILES / 4, 128, 0, stream>>>(x1, x2, lin_wt, linb, batch, out);
}
